// MyModel_Attn_18511309045965
// MI455X (gfx1250) — matrix operand order repaired and checked by reading, not compiled, not run on hardware
//
#include <hip/hip_runtime.h>
#include <hip/hip_bf16.h>

typedef __bf16 v16bf __attribute__((ext_vector_type(16)));
typedef __bf16 v8bf  __attribute__((ext_vector_type(8)));
typedef float  v8f   __attribute__((ext_vector_type(8)));

__device__ __forceinline__ v8f wmma_bf16(v16bf a, v16bf b, v8f c) {
    // D = A(16x32 bf16) * B(32x16 bf16) + C(16x16 f32)
    return __builtin_amdgcn_wmma_f32_16x16x32_bf16(false, a, false, b, (short)0, c, false, false);
}

// Load a 16-element bf16 fragment as two 16B vector loads (works for LDS or global).
__device__ __forceinline__ v16bf ld_frag(const __bf16* lo_p, const __bf16* hi_p) {
    v8bf lo = *(const v8bf*)lo_p;
    v8bf hi = *(const v8bf*)hi_p;
    return __builtin_shufflevector(lo, hi, 0, 1, 2, 3, 4, 5, 6, 7, 8, 9, 10, 11, 12, 13, 14, 15);
}

// CDNA5 async copy: 16 bytes global -> LDS per lane, tracked by ASYNCcnt.
__device__ __forceinline__ void async_copy_b128(void* lds_dst, const void* gsrc) {
    unsigned l = (unsigned)(unsigned long long)lds_dst;   // low 32 bits = LDS byte address
    asm volatile("global_load_async_to_lds_b128 %0, %1, off"
                 :: "v"(l), "v"(gsrc)
                 : "memory");
}
__device__ __forceinline__ void wait_async0() {
    asm volatile("s_wait_asynccnt 0" ::: "memory");
}
__device__ __forceinline__ void wait_async3() {
    // each thread keeps exactly 3 copies in flight per stage; in-order completion
    // means asynccnt<=3 guarantees the *previous* stage has fully landed.
    asm volatile("s_wait_asynccnt 3" ::: "memory");
}

// ---------------------------------------------------------------------------
// Generic GEMM: C[M,N] = A[M,K](bf16) * W[N,K](bf16)^T + bias1 + bias2
// Block 256 (8 waves), WG tile 64x128, wave tile 16x64, K step 32.
// Double-buffered async load-to-LDS staging: the next K-slice streams in behind
// the current slice's WMMAs (s_wait_asynccnt 3 steady state).
// Requires M % 64 == 0 (true for all call sites); N tiles masked at epilogue.
// ---------------------------------------------------------------------------
__global__ __launch_bounds__(256) void gemm_bf16_kernel(
    const __bf16* __restrict__ A, const __bf16* __restrict__ W,
    const float* __restrict__ bias1, const float* __restrict__ bias2,
    void* __restrict__ Cout, int M, int N, int K, int c_bf16)
{
    __shared__ __bf16 As[2][64][40];    // rows padded to 80B to spread LDS banks
    __shared__ __bf16 Bs[2][128][40];

    int tilesN = (N + 127) >> 7;
    int tm = blockIdx.x / tilesN;
    int tn = blockIdx.x % tilesN;
    int m0 = tm << 6, n0 = tn << 7;

    int tid  = threadIdx.x;
    int lane = tid & 31, wid = tid >> 5;
    int wm = wid & 3, wn = wid >> 2;
    int lm = lane & 15, lh = lane >> 4;

    // per-thread staging segments (16B = 8 bf16 each)
    int arow = tid >> 2, aseg = tid & 3;                    // As: 64 rows x 4 segs
    int brow0 = tid >> 1, bseg0 = (tid & 1) << 1;           // Bs: 128 rows x 4 segs, 2/thread
    int brow_c0 = (n0 + brow0 < N) ? (n0 + brow0) : (N - 1);

    const __bf16* agp = A + (size_t)(m0 + arow) * K + aseg * 8;
    const __bf16* wgp = W + (size_t)brow_c0 * K + bseg0 * 8;

    auto stage = [&](int buf, int kk) {
        async_copy_b128(&As[buf][arow][aseg * 8], agp + kk);
        async_copy_b128(&Bs[buf][brow0][bseg0 * 8], wgp + kk);
        async_copy_b128(&Bs[buf][brow0][(bseg0 + 2) * 8], wgp + kk + 16);
    };

    v8f acc[4];
    for (int t = 0; t < 4; ++t)
        for (int r = 0; r < 8; ++r) acc[t][r] = 0.f;

    int nk = K >> 5;
    stage(0, 0);
    for (int it = 0; it < nk; ++it) {
        int cur = it & 1;
        if (it + 1 < nk) {
            stage(cur ^ 1, (it + 1) << 5);
            wait_async3();          // current slice landed; next still in flight
        } else {
            wait_async0();
        }
        __syncthreads();

        int ar = (wm << 4) + lm, koff = lh << 3;
        v16bf af = ld_frag(&As[cur][ar][koff], &As[cur][ar][16 + koff]);

        v16bf bfv[4];
        int kb = lh << 4;
        for (int t = 0; t < 4; ++t) {
            int br = (wn << 6) + (t << 4) + lm;
            bfv[t] = ld_frag(&Bs[cur][br][koff], &Bs[cur][br][16 + koff]);
        }
        for (int t = 0; t < 4; ++t)
            acc[t] = wmma_bf16(af, bfv[t], acc[t]);
        __syncthreads();            // cur buffer free for re-issue at it+2
    }

    for (int t = 0; t < 4; ++t) {
        for (int r = 0; r < 8; ++r) {
            int grow = m0 + (wm << 4) + r + (lh << 3);
            int gcol = n0 + (wn << 6) + (t << 4) + lm;
            if (grow < M && gcol < N) {
                float v = acc[t][r];
                if (bias1) v += bias1[gcol];
                if (bias2) v += bias2[gcol];
                if (c_bf16) ((__bf16*)Cout)[(size_t)grow * N + gcol] = (__bf16)v;
                else        ((float*)Cout)[(size_t)grow * N + gcol] = v;
            }
        }
    }
}

// ---------------------------------------------------------------------------
// LSTM recurrence. Grid = B/16 (independent batch slices), block 512 (16 waves).
// g += h@Whh^T via bf16 WMMA; wave w owns matching i/f/g/o columns -> wave-local
// activations; h in LDS (bf16), c in registers. Writes h both f32 and bf16.
// ---------------------------------------------------------------------------
__global__ __launch_bounds__(512) void lstm_rec_kernel(
    const float* __restrict__ gates, const __bf16* __restrict__ whhb,
    float* __restrict__ hseq, __bf16* __restrict__ hseq_bf, int T)
{
    __shared__ __bf16 hB[16][272];

    int tid = threadIdx.x, lane = tid & 31, wid = tid >> 5;
    int lm = lane & 15, lh = lane >> 4;
    int b0 = blockIdx.x << 4;

    float creg[8];
    for (int r = 0; r < 8; ++r) creg[r] = 0.f;
    for (int i = tid; i < 16 * 272; i += 512) ((__bf16*)hB)[i] = (__bf16)0.f;
    __syncthreads();

    for (int t = 0; t < T; ++t) {
        v8f acc[4];
        for (int g = 0; g < 4; ++g)
            for (int r = 0; r < 8; ++r) acc[g][r] = 0.f;

        for (int ks = 0; ks < 8; ++ks) {
            int koff = lh << 3;
            v16bf af = ld_frag(&hB[lm][(ks << 5) + koff], &hB[lm][(ks << 5) + 16 + koff]);
            v16bf bfv[4];
            for (int g = 0; g < 4; ++g) {
                int wr = (g << 8) + (wid << 4) + lm;
                const __bf16* wp = whhb + (size_t)wr * 256 + (ks << 5) + (lh << 3);
                bfv[g] = ld_frag(wp, wp + 16);
            }
            for (int g = 0; g < 4; ++g)
                acc[g] = wmma_bf16(af, bfv[g], acc[g]);
        }
        __syncthreads();   // all waves done reading hB

        for (int r = 0; r < 8; ++r) {
            int lrow = r + (lh << 3);
            int gb   = b0 + lrow;
            int col  = (wid << 4) + lm;
            size_t gbase = ((size_t)gb * T + t) * 1024;
            __builtin_prefetch(&gates[gbase + 1024 + col], 0, 0);
            float iv = acc[0][r] + gates[gbase + col];
            float fv = acc[1][r] + gates[gbase + 256 + col];
            float gv = acc[2][r] + gates[gbase + 512 + col];
            float ov = acc[3][r] + gates[gbase + 768 + col];
            float si = 1.f / (1.f + __expf(-iv));
            float sf = 1.f / (1.f + __expf(-fv));
            float so = 1.f / (1.f + __expf(-ov));
            float cn = sf * creg[r] + si * tanhf(gv);
            creg[r] = cn;
            float hn = so * tanhf(cn);
            size_t oidx = ((size_t)gb * T + t) * 256 + col;
            hseq[oidx]    = hn;
            hseq_bf[oidx] = (__bf16)hn;
            hB[lrow][col] = (__bf16)hn;
        }
        __syncthreads();
    }
}

// ---------------------------------------------------------------------------
// Attention scores + softmax, writing directly transposed into d_out (NT stores).
// q,k are bf16. One WG per (b, h, 16-row s-tile).
// ---------------------------------------------------------------------------
__global__ __launch_bounds__(256) void attn_scores_kernel(
    const __bf16* __restrict__ q, const __bf16* __restrict__ k,
    float* __restrict__ attn_out)
{
    __shared__ float  sc[16][536];
    __shared__ __bf16 qs[16][72];
    __shared__ float  red[16][17];

    int tile = blockIdx.x;
    int st = tile % 33;
    int h  = (tile / 33) & 3;
    int b  = tile / (33 * 4);
    int s0 = st << 4;

    int tid = threadIdx.x, lane = tid & 31, wid = tid >> 5;
    int lm = lane & 15, lh = lane >> 4;

    // async stage q tile (16 x 64 bf16 = 2KB, 128 x 16B)
    if (tid < 128) {
        int r = tid >> 3, seg = tid & 7;
        int s = s0 + r; if (s > 519) s = 519;   // clamped rows are never written out
        async_copy_b128(&qs[r][seg * 8],
                        q + ((size_t)b * 520 + s) * 256 + h * 64 + seg * 8);
    }
    wait_async0();
    __syncthreads();

    v16bf af[2];
    for (int ks = 0; ks < 2; ++ks) {
        int koff = lh << 3;
        af[ks] = ld_frag(&qs[lm][(ks << 5) + koff], &qs[lm][(ks << 5) + 16 + koff]);
    }

    for (int jt = wid; jt < 33; jt += 8) {      // wave-uniform loop
        v8f acc;
        for (int r = 0; r < 8; ++r) acc[r] = 0.f;
        int tt = (jt << 4) + lm; if (tt > 519) tt = 519;  // cols >= 520 masked below
        for (int ks = 0; ks < 2; ++ks) {
            const __bf16* kp = k + ((size_t)b * 520 + tt) * 256 + h * 64 + (ks << 5) + (lh << 3);
            v16bf bfv = ld_frag(kp, kp + 16);
            acc = wmma_bf16(af[ks], bfv, acc);
        }
        for (int r = 0; r < 8; ++r) {
            int lrow = r + (lh << 3);
            int col  = (jt << 4) + lm;
            sc[lrow][col] = (col < 520) ? acc[r] * 0.125f : -3.0e38f;
        }
    }
    __syncthreads();

    // softmax: 16 threads per row
    int rr = tid >> 4, sub = tid & 15;
    float mx = -3.0e38f;
    for (int j = sub; j < 520; j += 16) mx = fmaxf(mx, sc[rr][j]);
    red[rr][sub] = mx;
    __syncthreads();
    if (sub == 0) {
        float m = red[rr][0];
        for (int j = 1; j < 16; ++j) m = fmaxf(m, red[rr][j]);
        red[rr][16] = m;
    }
    __syncthreads();
    float rowmax = red[rr][16];
    float sm = 0.f;
    for (int j = sub; j < 520; j += 16) {
        float e = __expf(sc[rr][j] - rowmax);
        sc[rr][j] = e;
        sm += e;
    }
    __syncthreads();
    red[rr][sub] = sm;
    __syncthreads();
    if (sub == 0) {
        float s = 0.f;
        for (int j = 0; j < 16; ++j) s += red[rr][j];
        red[rr][16] = s;
    }
    __syncthreads();
    float inv = 1.f / red[rr][16];
    int s = s0 + rr;
    if (s < 520) {
        float* op = attn_out + (((size_t)b * 520 + s) * 4 + h) * 520;
        for (int j = sub; j < 520; j += 16)
            __builtin_nontemporal_store(sc[rr][j] * inv, op + j);   // 277MB stream: TH=NT
    }
}

// head_sl[b,o,d] = mean_h sum_t attn[b, 512+o, h, t] * vs[b,t,d]  (bf16 out)
__global__ __launch_bounds__(64) void head_kernel(
    const float* __restrict__ attn, const float* __restrict__ vs,
    __bf16* __restrict__ head)
{
    int bo = blockIdx.x, b = bo >> 3, o = bo & 7, d = threadIdx.x;
    float a = 0.f;
    for (int h = 0; h < 4; ++h) {
        const float* ap = attn + (((size_t)b * 520 + 512 + o) * 4 + h) * 520;
        const float* vp = vs + (size_t)b * 520 * 64 + d;
        for (int t = 0; t < 520; ++t) a += ap[t] * vp[(size_t)t * 64];
    }
    head[(size_t)bo * 64 + d] = (__bf16)(a * 0.25f);
}

// concat(ts, sd) + layernorm over E=256 -> bf16 (consumed only by GEMMs)
__global__ __launch_bounds__(256) void ln_concat_kernel(
    const float* __restrict__ ts, const float* __restrict__ sd,
    const float* __restrict__ g, const float* __restrict__ be,
    __bf16* __restrict__ xn)
{
    __shared__ float s1[256], s2[256];
    int row = blockIdx.x, b = row / 520, s = row % 520, e = threadIdx.x;
    float v = (s < 512) ? ts[((size_t)b * 512 + s) * 256 + e]
                        : sd[((size_t)b * 8 + (s - 512)) * 256 + e];
    s1[e] = v; s2[e] = v * v;
    __syncthreads();
    for (int st = 128; st > 0; st >>= 1) {
        if (e < st) { s1[e] += s1[e + st]; s2[e] += s2[e + st]; }
        __syncthreads();
    }
    float mean = s1[0] * (1.f / 256.f);
    float var  = s2[0] * (1.f / 256.f) - mean * mean;
    xn[(size_t)row * 256 + e] = (__bf16)((v - mean) * rsqrtf(var + 1e-5f) * g[e] + be[e]);
}

__global__ __launch_bounds__(256) void embed_kernel(
    const float* __restrict__ x_ts, const float* __restrict__ W,
    const float* __restrict__ bi, __bf16* __restrict__ out, int n)
{
    int i = blockIdx.x * 256 + threadIdx.x;
    if (i < n) {
        int e = i & 255, bt = i >> 8;
        out[i] = (__bf16)(x_ts[bt] * W[e] + bi[e]);
    }
}

__global__ __launch_bounds__(256) void seed_gather_kernel(
    const int* __restrict__ idx, const float* __restrict__ tab,
    __bf16* __restrict__ out)
{
    int i = blockIdx.x * 256 + threadIdx.x;
    int e = i & 255, bo = i >> 8;
    out[i] = (__bf16)tab[(size_t)idx[bo] * 256 + e];
}

__global__ __launch_bounds__(256) void f32_to_bf16_kernel(
    const float* __restrict__ src, __bf16* __restrict__ dst, int n)
{
    int i = blockIdx.x * 256 + threadIdx.x;
    if (i < n) dst[i] = (__bf16)src[i];
}

__global__ __launch_bounds__(64) void tab_kernel(
    const float* __restrict__ x_tab,
    const float* __restrict__ W1, const float* __restrict__ b1,
    const float* __restrict__ W2, const float* __restrict__ b2,
    float* __restrict__ out)
{
    __shared__ float h1[64];
    int b = blockIdx.x, j = threadIdx.x;
    float a = b1[j];
    for (int c = 0; c < 32; ++c) a += x_tab[b * 32 + c] * W1[j * 32 + c];
    h1[j] = fmaxf(a, 0.f);
    __syncthreads();
    if (j < 32) {
        float a2 = b2[j];
        for (int c = 0; c < 64; ++c) a2 += h1[c] * W2[j * 64 + c];
        out[b * 32 + j] = fmaxf(a2, 0.f);
    }
}

__global__ __launch_bounds__(64) void mts_kernel(
    const float* __restrict__ osl,
    const float* __restrict__ W1, const float* __restrict__ b1,
    const float* __restrict__ W2, const float* __restrict__ b2,
    float* __restrict__ out)
{
    __shared__ float h1[64];
    int bo = blockIdx.x, o = bo & 7, f = threadIdx.x;
    float a = b1[o * 64 + f];
    const float* xp = osl + (size_t)bo * 256;
    const float* wp = W1 + ((size_t)o * 64 + f) * 256;
    for (int e = 0; e < 256; ++e) a += xp[e] * wp[e];
    h1[f] = fmaxf(a, 0.f);
    __syncthreads();
    if (f < 32) {
        float a2 = b2[o * 32 + f];
        const float* w2 = W2 + ((size_t)o * 32 + f) * 64;
        for (int c = 0; c < 64; ++c) a2 += h1[c] * w2[c];
        out[(size_t)bo * 32 + f] = fmaxf(a2, 0.f);
    }
}

__global__ __launch_bounds__(128) void reg_kernel(
    const float* __restrict__ out_tab, const float* __restrict__ out_ts,
    const float* __restrict__ g, const float* __restrict__ be,
    const float* __restrict__ W1, const float* __restrict__ b1,
    const float* __restrict__ W2, const float* __restrict__ b2,
    float* __restrict__ final_out)
{
    __shared__ float comb[64];
    __shared__ float rr[128];
    __shared__ float stat[2];
    int bo = blockIdx.x, b = bo >> 3, o = bo & 7, tid = threadIdx.x;
    if (tid < 32) comb[tid] = out_tab[b * 32 + tid];
    else if (tid < 64) comb[tid] = out_ts[bo * 32 + tid - 32];
    __syncthreads();
    if (tid == 0) {
        float s = 0.f, s2 = 0.f;
        for (int c = 0; c < 64; ++c) { float v = comb[c]; s += v; s2 += v * v; }
        float m = s * (1.f / 64.f);
        stat[0] = m;
        stat[1] = rsqrtf(s2 * (1.f / 64.f) - m * m + 1e-5f);
    }
    __syncthreads();
    if (tid < 64) comb[tid] = (comb[tid] - stat[0]) * stat[1] * g[o * 64 + tid] + be[o * 64 + tid];
    __syncthreads();
    float a = b1[o * 128 + tid];
    const float* wp = W1 + ((size_t)o * 128 + tid) * 64;
    for (int c = 0; c < 64; ++c) a += comb[c] * wp[c];
    a = fmaxf(a, 0.f) * W2[o * 128 + tid];
    rr[tid] = a;
    __syncthreads();
    for (int st = 64; st > 0; st >>= 1) {
        if (tid < st) rr[tid] += rr[tid + st];
        __syncthreads();
    }
    if (tid == 0) final_out[bo] = rr[0] + b2[o];
}

// ---------------------------------------------------------------------------
extern "C" void kernel_launch(void* const* d_in, const int* in_sizes, int n_in,
                              void* d_out, int out_size, void* d_ws, size_t ws_size,
                              hipStream_t stream)
{
    (void)in_sizes; (void)n_in; (void)out_size; (void)ws_size;
    const float* x_tab   = (const float*)d_in[0];
    const float* x_ts    = (const float*)d_in[1];
    const int*   x_seed  = (const int*)d_in[2];
    const float* emb_W   = (const float*)d_in[3];
    const float* emb_b   = (const float*)d_in[4];
    const float* l1_Wih  = (const float*)d_in[5];
    const float* l1_Whh  = (const float*)d_in[6];
    const float* l1_bih  = (const float*)d_in[7];
    const float* l1_bhh  = (const float*)d_in[8];
    const float* l2_Wih  = (const float*)d_in[9];
    const float* l2_Whh  = (const float*)d_in[10];
    const float* l2_bih  = (const float*)d_in[11];
    const float* l2_bhh  = (const float*)d_in[12];
    const float* seeder  = (const float*)d_in[13];
    const float* ln_g    = (const float*)d_in[14];
    const float* ln_b    = (const float*)d_in[15];
    const float* qW      = (const float*)d_in[16];
    const float* kW      = (const float*)d_in[17];
    const float* vW      = (const float*)d_in[18];
    const float* whW     = (const float*)d_in[19];
    const float* mts_W1  = (const float*)d_in[20];
    const float* mts_b1  = (const float*)d_in[21];
    const float* mts_W2  = (const float*)d_in[22];
    const float* mts_b2  = (const float*)d_in[23];
    const float* mtab_W1 = (const float*)d_in[24];
    const float* mtab_b1 = (const float*)d_in[25];
    const float* mtab_W2 = (const float*)d_in[26];
    const float* mtab_b2 = (const float*)d_in[27];
    const float* reg_g   = (const float*)d_in[28];
    const float* reg_b   = (const float*)d_in[29];
    const float* reg_W1  = (const float*)d_in[30];
    const float* reg_b1  = (const float*)d_in[31];
    const float* reg_W2  = (const float*)d_in[32];
    const float* reg_b2  = (const float*)d_in[33];
    float* out = (float*)d_out;

    char* p = (char*)d_ws;
    auto alloc = [&](size_t bytes) -> void* {
        void* r = (void*)p;
        p += (bytes + 255) & ~(size_t)255;
        return r;
    };
    // f32 buffers
    float*  gates = (float*)alloc(64ull * 512 * 1024 * 4);
    float*  ts_a  = (float*)alloc(64ull * 512 * 256 * 4);
    float*  ts_b  = (float*)alloc(64ull * 512 * 256 * 4);
    float*  sd_a  = (float*)alloc(64ull * 8 * 256 * 4);
    float*  sd_b  = (float*)alloc(64ull * 8 * 256 * 4);
    float*  sd_g  = (float*)alloc(64ull * 8 * 1024 * 4);
    float*  vsb   = (float*)alloc(64ull * 520 * 64 * 4);
    float*  osl   = (float*)alloc(512ull * 256 * 4);
    float*  otab  = (float*)alloc(64ull * 32 * 4);
    float*  ots   = (float*)alloc(512ull * 32 * 4);
    // bf16 buffers
    __bf16* tsbf_a = (__bf16*)alloc(64ull * 512 * 256 * 2);
    __bf16* tsbf_b = (__bf16*)alloc(64ull * 512 * 256 * 2);
    __bf16* sdbf_a = (__bf16*)alloc(64ull * 8 * 256 * 2);
    __bf16* sdbf_b = (__bf16*)alloc(64ull * 8 * 256 * 2);
    __bf16* xnbf   = (__bf16*)alloc(64ull * 520 * 256 * 2);
    __bf16* qbf    = (__bf16*)alloc(64ull * 520 * 256 * 2);
    __bf16* kbf    = (__bf16*)alloc(64ull * 520 * 256 * 2);
    __bf16* headbf = (__bf16*)alloc(512ull * 64 * 2);
    __bf16* whhb   = (__bf16*)alloc(1024ull * 256 * 2);
    __bf16* wih1   = (__bf16*)alloc(2ull * 1024 * 256 * 2);
    __bf16* wih2   = (__bf16*)alloc(2ull * 1024 * 256 * 2);
    __bf16* qWb    = (__bf16*)alloc(256ull * 256 * 2);
    __bf16* kWb    = (__bf16*)alloc(256ull * 256 * 2);
    __bf16* vWb    = (__bf16*)alloc(64ull * 256 * 2);
    __bf16* whWb   = (__bf16*)alloc(256ull * 64 * 2);

    // 0) one-time weight conversions to bf16
    f32_to_bf16_kernel<<<2048, 256, 0, stream>>>(l1_Wih, wih1, 2 * 1024 * 256);
    f32_to_bf16_kernel<<<2048, 256, 0, stream>>>(l2_Wih, wih2, 2 * 1024 * 256);
    f32_to_bf16_kernel<<<256, 256, 0, stream>>>(qW, qWb, 256 * 256);
    f32_to_bf16_kernel<<<256, 256, 0, stream>>>(kW, kWb, 256 * 256);
    f32_to_bf16_kernel<<<64, 256, 0, stream>>>(vW, vWb, 64 * 256);
    f32_to_bf16_kernel<<<64, 256, 0, stream>>>(whW, whWb, 256 * 64);

    // 1) embed time-series (bf16 out, consumed by gate GEMM)
    embed_kernel<<<(64 * 512 * 256) / 256, 256, 0, stream>>>(x_ts, emb_W, emb_b, tsbf_a, 64 * 512 * 256);

    // 2) LSTM stack 1 (T=512, 2 layers)
    {
        int M = 64 * 512, N = 1024, K = 256;
        int grid = (M / 64) * (N / 128);
        gemm_bf16_kernel<<<grid, 256, 0, stream>>>(tsbf_a, wih1, l1_bih, l1_bhh, gates, M, N, K, 0);
        f32_to_bf16_kernel<<<1024, 256, 0, stream>>>(l1_Whh, whhb, 1024 * 256);
        lstm_rec_kernel<<<4, 512, 0, stream>>>(gates, whhb, ts_b, tsbf_b, 512);

        gemm_bf16_kernel<<<grid, 256, 0, stream>>>(tsbf_b, wih1 + 1024 * 256, l1_bih + 1024, l1_bhh + 1024, gates, M, N, K, 0);
        f32_to_bf16_kernel<<<1024, 256, 0, stream>>>(l1_Whh + 1024 * 256, whhb, 1024 * 256);
        lstm_rec_kernel<<<4, 512, 0, stream>>>(gates, whhb, ts_a, tsbf_a, 512);
    }

    // 3) seeder LSTM stack (T=8, 2 layers)
    seed_gather_kernel<<<(64 * 8 * 256) / 256, 256, 0, stream>>>(x_seed, seeder, sdbf_a);
    {
        int M = 64 * 8, N = 1024, K = 256;
        int grid = (M / 64) * (N / 128);
        gemm_bf16_kernel<<<grid, 256, 0, stream>>>(sdbf_a, wih2, l2_bih, l2_bhh, sd_g, M, N, K, 0);
        f32_to_bf16_kernel<<<1024, 256, 0, stream>>>(l2_Whh, whhb, 1024 * 256);
        lstm_rec_kernel<<<4, 512, 0, stream>>>(sd_g, whhb, sd_b, sdbf_b, 8);

        gemm_bf16_kernel<<<grid, 256, 0, stream>>>(sdbf_b, wih2 + 1024 * 256, l2_bih + 1024, l2_bhh + 1024, sd_g, M, N, K, 0);
        f32_to_bf16_kernel<<<1024, 256, 0, stream>>>(l2_Whh + 1024 * 256, whhb, 1024 * 256);
        lstm_rec_kernel<<<4, 512, 0, stream>>>(sd_g, whhb, sd_a, sdbf_a, 8);
    }

    // 4) concat + layernorm -> bf16
    ln_concat_kernel<<<64 * 520, 256, 0, stream>>>(ts_a, sd_a, ln_g, ln_b, xnbf);

    // 5) q/k (bf16 out) and v (f32 out) projections
    {
        int M = 64 * 520, N = 256, K = 256;
        int grid = (M / 64) * (N / 128);
        gemm_bf16_kernel<<<grid, 256, 0, stream>>>(xnbf, qWb, nullptr, nullptr, qbf, M, N, K, 1);
        gemm_bf16_kernel<<<grid, 256, 0, stream>>>(xnbf, kWb, nullptr, nullptr, kbf, M, N, K, 1);
    }
    gemm_bf16_kernel<<<(64 * 520) / 64, 256, 0, stream>>>(xnbf, vWb, nullptr, nullptr, vsb, 64 * 520, 64, 256, 0);

    // 6) attention scores + softmax -> d_out (transposed), head for last 8 rows
    attn_scores_kernel<<<64 * 4 * 33, 256, 0, stream>>>(qbf, kbf, out + 512);
    head_kernel<<<512, 64, 0, stream>>>(out + 512, vsb, headbf);
    gemm_bf16_kernel<<<(512 / 64) * (256 / 128), 256, 0, stream>>>(headbf, whWb, nullptr, nullptr, osl, 512, 256, 64, 0);

    // 7) small heads
    tab_kernel<<<64, 64, 0, stream>>>(x_tab, mtab_W1, mtab_b1, mtab_W2, mtab_b2, otab);
    mts_kernel<<<512, 64, 0, stream>>>(osl, mts_W1, mts_b1, mts_W2, mts_b2, ots);
    reg_kernel<<<512, 128, 0, stream>>>(otab, ots, reg_g, reg_b, reg_W1, reg_b1, reg_W2, reg_b2, out);
}
